// MultiHeadAttentionLayer_75900662055078
// MI455X (gfx1250) — compile-verified, matrix operand order checked by reading, not run on hardware
//
#include <hip/hip_runtime.h>

#define BB 2
#define SS 2048
#define DD 1024
#define HH 16
#define DHH 64

typedef __attribute__((ext_vector_type(16))) __bf16 v16bf;
typedef __attribute__((ext_vector_type(8)))  float  v8f;
typedef unsigned short ush;

struct Frag16 { union { v16bf v; uint4 q[2]; }; };

// A/B fragment loader for v_wmma_f32_16x16x32_bf16.
// Per ISA layout: lane L holds row (L&15); K elements are two contiguous
// 8-element (16B) runs at k0 = 8*(L>>4) and k0+16.  Source must be
// row-major with `stride` bf16 elements between rows.
__device__ inline void load_frag(Frag16& f, const ush* base, int stride) {
  int lane = threadIdx.x & 31;
  const ush* p = base + (lane & 15) * stride + ((lane >> 4) << 3);
  f.q[0] = *(const uint4*)(p);
  f.q[1] = *(const uint4*)(p + 16);
}

__device__ inline v8f wmma_bf16(const Frag16& a, const Frag16& b, v8f c) {
  return __builtin_amdgcn_wmma_f32_16x16x32_bf16(
      /*neg_a=*/false, a.v, /*neg_b=*/false, b.v,
      /*c_mod=*/(short)0, c, /*reuse_a=*/false, /*reuse_b=*/false);
}

__device__ inline ush f2bf(float x) {  // round-to-nearest-even f32 -> bf16 bits
  union { float f; unsigned u; } v; v.f = x;
  unsigned r = v.u + 0x7FFFu + ((v.u >> 16) & 1u);
  return (ush)(r >> 16);
}

__device__ inline float rmax16(float v) {
  #pragma unroll
  for (int i = 1; i < 16; i <<= 1) v = fmaxf(v, __shfl_xor(v, i, 16));
  return v;
}
__device__ inline float rsum16(float v) {
  #pragma unroll
  for (int i = 1; i < 16; i <<= 1) v += __shfl_xor(v, i, 16);
  return v;
}

// ---------------- prep kernels ----------------
__global__ void cvt_x_kernel(const float* __restrict__ X, ush* __restrict__ Xbf, int n) {
  for (int i = blockIdx.x * blockDim.x + threadIdx.x; i < n; i += gridDim.x * blockDim.x)
    Xbf[i] = f2bf(X[i]);
}

// WT layout: [3][H][DH][D] bf16  (row e contiguous over d) — B-fragment friendly
__global__ void cvt_w_kernel(const float* __restrict__ Wq, const float* __restrict__ Wk,
                             const float* __restrict__ Wv, ush* __restrict__ WT, int n) {
  const int HDD = HH * DHH * DD;
  for (int i = blockIdx.x * blockDim.x + threadIdx.x; i < n; i += gridDim.x * blockDim.x) {
    int m = i / HDD;
    int r = i - m * HDD;
    int h = r / (DHH * DD);
    int r2 = r - h * (DHH * DD);
    int e = r2 / DD;
    int d = r2 - e * DD;
    const float* W = (m == 0) ? Wq : (m == 1) ? Wk : Wv;
    WT[i] = f2bf(W[(h * DD + d) * DHH + e]);
  }
}

// ---------------- QKV projection (software-pipelined) ----------------
// One wave = 16 rows x 64 cols of one (mt,h).  Two ping-pong fragment sets so
// next k-step's 10 b128 loads are in flight while the current 4 WMMAs execute.
struct FragSet { Frag16 a; Frag16 b[4]; };

__device__ inline void load_set(FragSet& s, const ush* Abase, const ush* Bbase, int kk) {
  load_frag(s.a, Abase + kk, DD);
  #pragma unroll
  for (int t = 0; t < 4; ++t)
    load_frag(s.b[t], Bbase + (size_t)(t * 16) * DD + kk, DD);
}

__global__ __launch_bounds__(32) void qkv_proj_kernel(
    const ush* __restrict__ Xbf, const ush* __restrict__ WT,
    const float* __restrict__ bq, const float* __restrict__ bk, const float* __restrict__ bv,
    ush* __restrict__ Qbf, ush* __restrict__ Kbf, ush* __restrict__ Vt) {
  int wid = blockIdx.x;                // 0 .. 3*16*256-1
  int mt  = wid / (HH * 256);          // 0=q 1=k 2=v
  int rem = wid - mt * (HH * 256);
  int h   = rem >> 8;
  int rt  = rem & 255;
  int m0  = rt << 4;                   // flat row in [0, B*S)
  int b   = m0 >> 11;
  int s0  = m0 & (SS - 1);

  const ush* Abase = Xbf + (size_t)m0 * DD;
  const ush* Bbase = WT + ((size_t)(mt * HH + h) * DHH) * DD;

  v8f acc[4] = {};
  FragSet f0, f1;
  load_set(f0, Abase, Bbase, 0);
  for (int kk = 0; kk < DD; kk += 64) {
    load_set(f1, Abase, Bbase, kk + 32);      // in flight during f0's WMMAs
    #pragma unroll
    for (int t = 0; t < 4; ++t) acc[t] = wmma_bf16(f0.a, f0.b[t], acc[t]);
    if (kk + 64 < DD)
      load_set(f0, Abase, Bbase, kk + 64);    // in flight during f1's WMMAs
    #pragma unroll
    for (int t = 0; t < 4; ++t) acc[t] = wmma_bf16(f1.a, f1.b[t], acc[t]);
  }

  int lane = threadIdx.x & 31, col = lane & 15, hi = lane >> 4;
  const float* bp = (mt == 0) ? bq : (mt == 1) ? bk : bv;
  const float cQ = 0.125f * 1.44269504088896340736f;  // 1/sqrt(DH) * log2(e) folded into Q
  #pragma unroll
  for (int t = 0; t < 4; ++t) {
    #pragma unroll
    for (int r = 0; r < 8; ++r) {
      int e   = t * 16 + col;
      int row = r + (hi << 3);
      int s   = s0 + row;
      float val = acc[t][r] + bp[h * DHH + e];
      if (mt == 0) {
        Qbf[((size_t)(b * HH + h) * SS + s) * DHH + e] = f2bf(val * cQ);
      } else if (mt == 1) {
        Kbf[((size_t)(b * HH + h) * SS + s) * DHH + e] = f2bf(val);
      } else {  // V stored transposed [e][s] so P@V B-fragments are contiguous
        Vt[((size_t)(b * HH + h) * DHH + e) * SS + s] = f2bf(val);
      }
    }
  }
}

// K fragments for one 32-wide kv step: cols kv0..+15 / kv0+16..+31, e halves 0..31 / 32..63
__device__ inline void loadK4(Frag16* kf, const ush* Kb, int kv0) {
  load_frag(kf[0], Kb + (size_t)kv0 * DHH, DHH);
  load_frag(kf[1], Kb + (size_t)kv0 * DHH + 32, DHH);
  load_frag(kf[2], Kb + (size_t)(kv0 + 16) * DHH, DHH);
  load_frag(kf[3], Kb + (size_t)(kv0 + 16) * DHH + 32, DHH);
}

// ---------------- fused attention: one wave = one (b,h) 16-row q tile ----------------
__global__ __launch_bounds__(32) void attn_kernel(
    const ush* __restrict__ Qbf, const ush* __restrict__ Kbf,
    const ush* __restrict__ Vt, float* __restrict__ out) {
  __shared__ ush ldsP[16 * 32];   // P tile bounce buffer (C-layout -> A-layout)

  int wid = blockIdx.x;           // 0 .. B*H*128-1
  int b   = wid >> 11;
  int h   = (wid >> 7) & (HH - 1);
  int qt  = wid & 127;
  int qBase = qt << 4;

  const ush* Qb = Qbf + ((size_t)(b * HH + h) * SS + qBase) * DHH;
  const ush* Kb = Kbf + (size_t)(b * HH + h) * SS * DHH;
  const ush* Vb = Vt  + (size_t)(b * HH + h) * DHH * SS;

  Frag16 qf0, qf1;                // resident for the whole kv loop
  load_frag(qf0, Qb, DHH);        // e = 0..31
  load_frag(qf1, Qb + 32, DHH);   // e = 32..63

  int lane = threadIdx.x & 31, col = lane & 15, hi = lane >> 4;
  float m[8], l[8];
  v8f o0 = {}, o1 = {}, o2 = {}, o3 = {};
  #pragma unroll
  for (int r = 0; r < 8; ++r) { m[r] = -INFINITY; l[r] = 0.0f; }

  Frag16 kf[4];
  loadK4(kf, Kb, 0);

  for (int kv0 = 0; kv0 < SS; kv0 += 32) {
    bool active = (kv0 < qBase + 16);  // any unmasked column in this tile?

    // Issue V loads first: they have the score GEMM + softmax window to land.
    Frag16 vf[4];
    if (active) {
      load_frag(vf[0], Vb + (size_t)0  * SS + kv0, SS);
      load_frag(vf[1], Vb + (size_t)16 * SS + kv0, SS);
      load_frag(vf[2], Vb + (size_t)32 * SS + kv0, SS);
      load_frag(vf[3], Vb + (size_t)48 * SS + kv0, SS);
    }
    if (kv0 + 32 < SS)  // warm next K tile toward L2/L0
      __builtin_prefetch(Kb + (size_t)(kv0 + 32) * DHH + lane * 16, 0, 0);

    // scores: 16 q rows x 32 kv cols, accumulated over e=0..63
    v8f s0v = {}, s1v = {};
    s0v = wmma_bf16(qf0, kf[0], s0v);
    s0v = wmma_bf16(qf1, kf[1], s0v);
    s1v = wmma_bf16(qf0, kf[2], s1v);
    s1v = wmma_bf16(qf1, kf[3], s1v);

    // Next iteration's K fragments: hidden behind the softmax VALU block below.
    if (kv0 + 32 < SS) loadK4(kf, Kb, kv0 + 32);

    // online softmax: statistics over the FULL row (reference semantics),
    // mask applied only to the V-accumulation numerator.
    #pragma unroll
    for (int r = 0; r < 8; ++r) {
      float rm   = rmax16(fmaxf(s0v[r], s1v[r]));
      float mnew = fmaxf(m[r], rm);
      float sc   = exp2f(m[r] - mnew);
      float p0   = exp2f(s0v[r] - mnew);
      float p1   = exp2f(s1v[r] - mnew);
      l[r] = l[r] * sc + rsum16(p0 + p1);
      m[r] = mnew;
      o0[r] *= sc; o1[r] *= sc; o2[r] *= sc; o3[r] *= sc;
      if (active) {
        int row = r + (hi << 3);
        int qg  = qBase + row;
        ldsP[row * 32 + col]      = (kv0 + col      <= qg) ? f2bf(p0) : (ush)0;
        ldsP[row * 32 + col + 16] = (kv0 + 16 + col <= qg) ? f2bf(p1) : (ush)0;
      }
    }

    if (active) {  // P @ V over this 32-wide kv chunk (skipped when fully masked)
      Frag16 pf;
      const ush* pp = &ldsP[(lane & 15) * 32 + ((lane >> 4) << 3)];
      pf.q[0] = *(const uint4*)(pp);
      pf.q[1] = *(const uint4*)(pp + 16);
      o0 = wmma_bf16(pf, vf[0], o0);
      o1 = wmma_bf16(pf, vf[1], o1);
      o2 = wmma_bf16(pf, vf[2], o2);
      o3 = wmma_bf16(pf, vf[3], o3);
    }
  }

  // divide masked numerator by full-row softmax sum; write [B,S,H*DH]
  #pragma unroll
  for (int r = 0; r < 8; ++r) {
    float inv = 1.0f / l[r];
    int row = r + (hi << 3);
    float* op = out + ((size_t)(b * SS + qBase + row)) * DD + h * DHH + col;
    op[0]  = o0[r] * inv;
    op[16] = o1[r] * inv;
    op[32] = o2[r] * inv;
    op[48] = o3[r] * inv;
  }
}

extern "C" void kernel_launch(void* const* d_in, const int* in_sizes, int n_in,
                              void* d_out, int out_size, void* d_ws, size_t ws_size,
                              hipStream_t stream) {
  (void)in_sizes; (void)n_in; (void)out_size; (void)ws_size;
  const float* X  = (const float*)d_in[0];
  const float* Wq = (const float*)d_in[1];
  const float* Wk = (const float*)d_in[2];
  const float* Wv = (const float*)d_in[3];
  const float* bq = (const float*)d_in[4];
  const float* bk = (const float*)d_in[5];
  const float* bv = (const float*)d_in[6];
  float* out = (float*)d_out;

  char* ws = (char*)d_ws;
  ush* Xbf = (ush*)(ws);                         //  8 MB  [B*S, D]
  ush* WT  = (ush*)(ws + 8388608);               //  6 MB  [3,H,DH,D]
  ush* Qbf = (ush*)(ws + 14680064);              //  8 MB  [B,H,S,DH] (pre-scaled)
  ush* Kbf = (ush*)(ws + 23068672);              //  8 MB  [B,H,S,DH]
  ush* Vt  = (ush*)(ws + 31457280);              //  8 MB  [B,H,DH,S] (transposed)

  cvt_x_kernel<<<2048, 256, 0, stream>>>(X, Xbf, BB * SS * DD);
  cvt_w_kernel<<<2048, 256, 0, stream>>>(Wq, Wk, Wv, WT, 3 * HH * DHH * DD);
  qkv_proj_kernel<<<3 * HH * 256, 32, 0, stream>>>(Xbf, WT, bq, bk, bv, Qbf, Kbf, Vt);
  attn_kernel<<<BB * HH * 128, 32, 0, stream>>>(Qbf, Kbf, Vt, out);
}
